// MultiheadL2Attention_42795054137395
// MI455X (gfx1250) — matrix operand order repaired and checked by reading, not compiled, not run on hardware
//
#include <hip/hip_runtime.h>
#include <math.h>
#include <stdint.h>

#define B_ 2
#define L_ 2048
#define E_ 1024
#define H_ 16
#define D_ 64

typedef __attribute__((ext_vector_type(16))) __bf16 v16bf;
typedef __attribute__((ext_vector_type(8))) float v8f;
typedef __attribute__((ext_vector_type(4))) unsigned int v4u_;
typedef __attribute__((ext_vector_type(8))) int v8i_;
typedef __attribute__((ext_vector_type(4))) int v4i_;

union BFrag {
  v16bf bf;
  unsigned int u[8];
  unsigned short us[16];
};

// hardware bf16 converts (v_cvt_*_bf16) instead of software RNE bit math
__device__ __forceinline__ unsigned short f2bf(float f) {
  __bf16 h = (__bf16)f;
  return __builtin_bit_cast(unsigned short, h);
}
__device__ __forceinline__ float bf2f(unsigned short u) {
  __bf16 h = __builtin_bit_cast(__bf16, u);
  return (float)h;
}

// ---------------------------------------------------------------------------
// TDM: async DMA of a 2-D tile (tile_dim0=32 elems of 2B contiguous,
// tile_dim1=64 rows, row stride = L_ elements) from global into LDS.
// D# built per CDNA5 ISA ch.8 (group0: count/lds/global/type, group1: dims).
// Groups 2/3 zero: tensor is 2-D (tile_dim2=0 -> unused).
// This toolchain's builtin is the 6-arg form:
//   (uint32x4 g0, int32x8 g1, int32x4, int32x4, int32x8, i32 cpol)
// ---------------------------------------------------------------------------
__device__ __forceinline__ void tdm_load_v_tile(unsigned lds_off,
                                                const unsigned short* gptr) {
  unsigned long long ga = (unsigned long long)(uintptr_t)gptr;
  v4u_ g0;
  g0[0] = 1u;                                   // count=1 (valid), user mode
  g0[1] = lds_off;                              // lds_addr (bytes)
  g0[2] = (unsigned)(ga & 0xFFFFFFFFu);         // global_addr[31:0]
  g0[3] = (unsigned)((ga >> 32) & 0x01FFFFFFu)  // global_addr[56:32]
          | (2u << 30);                         // type=2 ("image")
  v8i_ g1;
  g1[0] = (int)(1u << 16);               // workgroup_mask=0, data_size=1 (2B)
  g1[1] = (int)((2048u & 0xFFFFu) << 16);       // tensor_dim0[15:0] @bit48
  g1[2] = (int)((2048u >> 16) | (64u << 16));   // dim0 hi | tensor_dim1 lo
  g1[3] = (int)(0u | (32u << 16));              // dim1 hi | tile_dim0=32
  g1[4] = (int)(64u);                           // tile_dim1=64, tile_dim2=0
  g1[5] = (int)(2048u);                         // tensor_dim0_stride[31:0]
  g1[6] = 0;                                    // stride hi | dim1_stride lo
  g1[7] = 0;
  v4i_ gz4 = {0, 0, 0, 0};
  v8i_ gz8 = {0, 0, 0, 0, 0, 0, 0, 0};
  __builtin_amdgcn_tensor_load_to_lds(g0, g1, gz4, gz4, gz8, 0);
}

// ---------------------------------------------------------------------------
// Projection: q = X @ W^T per head tile. One wave computes a 16x64 tile.
// A-frag (16x32 bf16): lane<16 row=lane&15, K in {0..7,16..23}; lane>=16 K+8.
// B-frag (32x16 bf16): lane col = lane&15, same K order as A-frag.
// ---------------------------------------------------------------------------
__global__ __launch_bounds__(32) void proj_kernel(
    const float* __restrict__ Q, const float* __restrict__ K,
    const float* __restrict__ V, const float* __restrict__ Wq,
    const float* __restrict__ Wk, const float* __restrict__ Wv,
    unsigned short* __restrict__ qbf, unsigned short* __restrict__ kbf,
    unsigned short* __restrict__ vbfT, float* __restrict__ q2,
    float* __restrict__ k2) {
  const int z = blockIdx.z;
  const float* X = (z == 0) ? Q : (z == 1) ? K : V;
  const float* W = (z == 0) ? Wq : (z == 1) ? Wk : Wv;
  const int lane = threadIdx.x & 31;
  const int half = lane >> 4;
  const int n16 = lane & 15;
  const int rowg0 = blockIdx.x * 16;  // flat row in (B*L); never straddles b
  const int b = rowg0 / L_;
  const int l0 = rowg0 % L_;
  const int h = blockIdx.y;
  const int ncol0 = h * D_;

  v8f acc[4] = {};

  const float* xrow = X + (size_t)(rowg0 + n16) * E_;
#pragma unroll 1
  for (int e0 = 0; e0 < E_; e0 += 32) {
    BFrag a;
#pragma unroll
    for (int v = 0; v < 8; ++v) {
      int kk = 2 * (v & 3) + 16 * (v >> 2) + 8 * half;
      float2 f = *(const float2*)(xrow + e0 + kk);
      a.bf[2 * v] = (__bf16)f.x;
      a.bf[2 * v + 1] = (__bf16)f.y;
    }
#pragma unroll
    for (int j = 0; j < 4; ++j) {
      BFrag bb;
      const float* wrow = W + (size_t)(ncol0 + j * 16 + n16) * E_;
#pragma unroll
      for (int v = 0; v < 8; ++v) {
        int kk = 2 * (v & 3) + 16 * (v >> 2) + 8 * half;
        float2 f = *(const float2*)(wrow + e0 + kk);
        bb.bf[2 * v] = (__bf16)f.x;
        bb.bf[2 * v + 1] = (__bf16)f.y;
      }
      acc[j] = __builtin_amdgcn_wmma_f32_16x16x32_bf16(
          false, a.bf, false, bb.bf, (short)0, acc[j], false, false);
    }
  }

  // C layout: VGPR r: lanes 0-15 -> row r, lanes 16-31 -> row 8+r; col=lane&15
  const int bh = b * H_ + h;
  if (z == 2) {
    // store V transposed: vbfT[(bh*D + d)*L + l] so PV B-frags get contiguous K
#pragma unroll
    for (int j = 0; j < 4; ++j) {
      int d = j * 16 + n16;
      unsigned int pk[4];
#pragma unroll
      for (int r = 0; r < 4; ++r) {
        unsigned short lo = f2bf(acc[j][2 * r]);
        unsigned short hi = f2bf(acc[j][2 * r + 1]);
        pk[r] = (unsigned)lo | ((unsigned)hi << 16);
      }
      unsigned short* dst =
          vbfT + ((size_t)(bh * D_ + d)) * L_ + l0 + 8 * half;
      *(uint4*)dst = make_uint4(pk[0], pk[1], pk[2], pk[3]);
    }
  } else {
    unsigned short* outb = (z == 0) ? qbf : kbf;
    float* sqout = (z == 0) ? q2 : k2;
    float ss[8] = {0.f, 0.f, 0.f, 0.f, 0.f, 0.f, 0.f, 0.f};
#pragma unroll
    for (int j = 0; j < 4; ++j) {
      int d = j * 16 + n16;
#pragma unroll
      for (int r = 0; r < 8; ++r) {
        unsigned short hb = f2bf(acc[j][r]);
        int l = l0 + 8 * half + r;
        outb[((size_t)bh * L_ + l) * D_ + d] = hb;
        float qv = bf2f(hb);
        ss[r] += qv * qv;
      }
    }
#pragma unroll
    for (int r = 0; r < 8; ++r) {
      float s = ss[r];
      s += __shfl_xor(s, 1, 32);
      s += __shfl_xor(s, 2, 32);
      s += __shfl_xor(s, 4, 32);
      s += __shfl_xor(s, 8, 32);
      ss[r] = s;
    }
    if (n16 == 0) {
#pragma unroll
      for (int r = 0; r < 8; ++r)
        sqout[(size_t)bh * L_ + l0 + 8 * half + r] = ss[r];
    }
  }
}

// ---------------------------------------------------------------------------
// Flash-style L2 attention: one wave = 16 query rows of one (b,h).
// Online softmax over 32-key blocks. V tiles are streamed by the Tensor Data
// Mover into double-buffered LDS (async, TENSORcnt), P is staged through LDS
// to build the bf16 A-fragment for the P@V WMMAs.
// ---------------------------------------------------------------------------
__global__ __launch_bounds__(32) void attn_kernel(
    const unsigned short* __restrict__ qbf,
    const unsigned short* __restrict__ kbf,
    const unsigned short* __restrict__ vbfT, const float* __restrict__ q2a,
    const float* __restrict__ k2a, float* __restrict__ ho) {
  __shared__ unsigned short ldsP[16 * 32];     // P tile (16x32 bf16)
  __shared__ unsigned short ldsV[2][64 * 32];  // double-buffered V tiles
  const int lane = threadIdx.x & 31;
  const int half = lane >> 4;
  const int n16 = lane & 15;
  const int qt = blockIdx.x;
  const int h = blockIdx.y;
  const int b = blockIdx.z;
  const int bh = b * H_ + h;
  const int q0 = qt * 16;

  const unsigned short* qbase = qbf + (size_t)bh * L_ * D_;
  const unsigned short* kbase = kbf + (size_t)bh * L_ * D_;
  const unsigned short* vbase = vbfT + (size_t)bh * D_ * L_;

  const unsigned ldsV_off0 = (unsigned)(uintptr_t)(&ldsV[0][0]);
  const unsigned ldsV_off1 = (unsigned)(uintptr_t)(&ldsV[1][0]);

  // Q A-fragments for feature halves 0..31 and 32..63 (kept in registers)
  BFrag aq[2];
#pragma unroll
  for (int f = 0; f < 2; ++f)
#pragma unroll
    for (int v = 0; v < 8; ++v) {
      int kk = 2 * (v & 3) + 16 * (v >> 2) + 8 * half;
      aq[f].u[v] =
          *(const unsigned int*)(qbase + (size_t)(q0 + n16) * D_ + 32 * f + kk);
    }

  float q2v[8];
#pragma unroll
  for (int r = 0; r < 8; ++r)
    q2v[r] = q2a[(size_t)bh * L_ + q0 + 8 * half + r];

  float mrun[8], zr[8];
#pragma unroll
  for (int r = 0; r < 8; ++r) {
    mrun[r] = -1e30f;
    zr[r] = 0.f;
  }
  v8f acc[4] = {};

  // prime the V pipeline: async TDM load of the first 64x32 tile
  tdm_load_v_tile(ldsV_off0, vbase + 0);

#pragma unroll 1
  for (int k0 = 0; k0 < L_; k0 += 32) {
    const int buf = (k0 >> 5) & 1;
    const bool more = (k0 + 32 < L_);
    if (more) {
      // async prefetch of the next V tile into the other buffer
      tdm_load_v_tile(buf ? ldsV_off0 : ldsV_off1, vbase + (k0 + 32));
      __builtin_prefetch(kbase + (size_t)(k0 + 32 + n16) * D_, 0, 1);
    }

    v8f S[2];
    float kk2[2];
#pragma unroll
    for (int s = 0; s < 2; ++s) {
      int keyrow = k0 + 16 * s + n16;
      BFrag kb0, kb1;
#pragma unroll
      for (int v = 0; v < 8; ++v) {
        int kk = 2 * (v & 3) + 16 * (v >> 2) + 8 * half;
        kb0.u[v] = *(const unsigned int*)(kbase + (size_t)keyrow * D_ + kk);
        kb1.u[v] =
            *(const unsigned int*)(kbase + (size_t)keyrow * D_ + 32 + kk);
      }
      v8f c = {};
      c = __builtin_amdgcn_wmma_f32_16x16x32_bf16(false, aq[0].bf, false,
                                                  kb0.bf, (short)0, c, false,
                                                  false);
      c = __builtin_amdgcn_wmma_f32_16x16x32_bf16(false, aq[1].bf, false,
                                                  kb1.bf, (short)0, c, false,
                                                  false);
      S[s] = c;
      kk2[s] = k2a[(size_t)bh * L_ + keyrow];
    }

    // scores = -sqrt(max(q2 + k2 - 2 q.k, 0)); online max / rescale
    float sc[2][8], tmax[8];
#pragma unroll
    for (int r = 0; r < 8; ++r) {
      float tm = -1e30f;
#pragma unroll
      for (int s = 0; s < 2; ++s) {
        float d2 = q2v[r] + kk2[s] - 2.f * S[s][r];
        d2 = fmaxf(d2, 0.f);
        float sco = -sqrtf(d2);
        sc[s][r] = sco;
        tm = fmaxf(tm, sco);
      }
      tmax[r] = tm;
    }
#pragma unroll
    for (int r = 0; r < 8; ++r) {
      float tm = tmax[r];
      tm = fmaxf(tm, __shfl_xor(tm, 1, 32));
      tm = fmaxf(tm, __shfl_xor(tm, 2, 32));
      tm = fmaxf(tm, __shfl_xor(tm, 4, 32));
      tm = fmaxf(tm, __shfl_xor(tm, 8, 32));
      float newm = fmaxf(mrun[r], tm);
      float scale = __expf(mrun[r] - newm);
      mrun[r] = newm;
      zr[r] *= scale;
#pragma unroll
      for (int j = 0; j < 4; ++j) acc[j][r] *= scale;
    }
#pragma unroll
    for (int s = 0; s < 2; ++s)
#pragma unroll
      for (int r = 0; r < 8; ++r) {
        float p = __expf(sc[s][r] - mrun[r]);
        zr[r] += p;
        ldsP[(r + 8 * half) * 32 + 16 * s + n16] = f2bf(p);
      }
    __syncthreads();

    // P as A-fragment from LDS
    BFrag ap;
#pragma unroll
    for (int v = 0; v < 8; ++v) {
      int kk = 2 * (v & 3) + 16 * (v >> 2) + 8 * half;
      ap.u[v] = *(const unsigned int*)&ldsP[n16 * 32 + kk];
    }

    // ensure the current V tile has landed (<=1 outstanding newer TDM op;
    // TENSORcnt is in-order per wave)
    if (more)
      __builtin_amdgcn_s_wait_tensorcnt((short)1);
    else
      __builtin_amdgcn_s_wait_tensorcnt((short)0);

    // V B-fragments from the LDS tile: row d holds 32 contiguous bf16 keys
#pragma unroll
    for (int j = 0; j < 4; ++j) {
      BFrag vb;
#pragma unroll
      for (int v = 0; v < 8; ++v) {
        int kk = 2 * (v & 3) + 16 * (v >> 2) + 8 * half;
        vb.u[v] =
            *(const unsigned int*)&ldsV[buf][(j * 16 + n16) * 32 + kk];
      }
      acc[j] = __builtin_amdgcn_wmma_f32_16x16x32_bf16(
          false, ap.bf, false, vb.bf, (short)0, acc[j], false, false);
    }
    __syncthreads();
  }

  // finalize: Z per row, sum O/Z over this wave's 16 query rows, atomic add
#pragma unroll
  for (int r = 0; r < 8; ++r) {
    float s = zr[r];
    s += __shfl_xor(s, 1, 32);
    s += __shfl_xor(s, 2, 32);
    s += __shfl_xor(s, 4, 32);
    s += __shfl_xor(s, 8, 32);
    zr[r] = s;
  }
#pragma unroll
  for (int j = 0; j < 4; ++j) {
    float cs = 0.f;
#pragma unroll
    for (int r = 0; r < 8; ++r) cs += acc[j][r] / zr[r];
    cs += __shfl_xor(cs, 16, 32);  // rows 0-7 + rows 8-15
    if (half == 0) atomicAdd(&ho[bh * D_ + j * 16 + n16], cs);
  }
}

__global__ void zero_kernel(float* p, int n) {
  int i = blockIdx.x * blockDim.x + threadIdx.x;
  if (i < n) p[i] = 0.f;
}

// out[b,n] = sum_e ho[b,e] * Wo[n,e]  (tiny GEMV, VALU is fine)
__global__ void out_gemv(const float* __restrict__ ho,
                         const float* __restrict__ Wo,
                         float* __restrict__ out) {
  int n = blockIdx.x * blockDim.x + threadIdx.x;
  int b = blockIdx.y;
  const float* hb = ho + b * E_;
  const float* wr = Wo + (size_t)n * E_;
  float s = 0.f;
  for (int e = 0; e < E_; e += 4) {
    float4 a = *(const float4*)(hb + e);
    float4 w = *(const float4*)(wr + e);
    s += a.x * w.x + a.y * w.y + a.z * w.z + a.w * w.w;
  }
  out[b * E_ + n] = s;
}

extern "C" void kernel_launch(void* const* d_in, const int* in_sizes, int n_in,
                              void* d_out, int out_size, void* d_ws,
                              size_t ws_size, hipStream_t stream) {
  (void)in_sizes; (void)n_in; (void)out_size; (void)ws_size;
  const float* Q = (const float*)d_in[0];
  const float* K = (const float*)d_in[1];
  const float* V = (const float*)d_in[2];
  const float* Wq = (const float*)d_in[3];
  const float* Wk = (const float*)d_in[4];
  const float* Wv = (const float*)d_in[5];
  const float* Wo = (const float*)d_in[6];
  float* out = (float*)d_out;

  // workspace: qbf/kbf/vbfT (bf16, 8MB each) + q2/k2 (256KB each) + ho (8KB)
  const size_t NQKV = (size_t)B_ * H_ * L_ * D_;
  unsigned short* qbf = (unsigned short*)d_ws;
  unsigned short* kbf = qbf + NQKV;
  unsigned short* vbfT = kbf + NQKV;
  float* q2 = (float*)(vbfT + NQKV);
  float* k2 = q2 + (size_t)B_ * H_ * L_;
  float* ho = k2 + (size_t)B_ * H_ * L_;

  zero_kernel<<<dim3((B_ * E_ + 255) / 256), 256, 0, stream>>>(ho, B_ * E_);
  proj_kernel<<<dim3((B_ * L_) / 16, H_, 3), 32, 0, stream>>>(
      Q, K, V, Wq, Wk, Wv, qbf, kbf, vbfT, q2, k2);
  attn_kernel<<<dim3(L_ / 16, H_, B_), 32, 0, stream>>>(qbf, kbf, vbfT, q2, k2,
                                                        ho);
  out_gemv<<<dim3(E_ / 256, B_), 256, 0, stream>>>(ho, Wo, out);
}
